// GATEncoder_3848290697595
// MI455X (gfx1250) — matrix operand order repaired and checked by reading, not compiled, not run on hardware
//
#include <hip/hip_runtime.h>
#include <hip/hip_bf16.h>

#define N_NODES  50000
#define E_EDGES  800000
#define ET       (E_EDGES + N_NODES)   // edges + self loops
#define IN_CH    128
#define HEADS    4
#define CH       64
#define HC       256
#define G_GRAPHS 64
#define NEG_SLOPE 0.2f
#define BN_EPS    1e-5f
#define ENC_NEG_INF 0x007FFFFFu        // encode(-inf) for uint-ordered float max

typedef __attribute__((ext_vector_type(16))) __bf16 v16bf;
typedef __attribute__((ext_vector_type(8)))  __bf16 v8bf;
typedef __attribute__((ext_vector_type(8)))  float  v8f;
typedef __attribute__((ext_vector_type(4)))  float  v4f;

// ---------------- utility kernels ----------------

__global__ void k_fill_f32(float* p, float v, int n) {
  int i = blockIdx.x * blockDim.x + threadIdx.x;
  if (i < n) p[i] = v;
}

__global__ void k_fill_u32(unsigned* p, unsigned v, int n) {
  int i = blockIdx.x * blockDim.x + threadIdx.x;
  if (i < n) p[i] = v;
}

__global__ void k_f32_to_bf16(const float* __restrict__ s, __bf16* __restrict__ d, int n) {
  int i = blockIdx.x * blockDim.x + threadIdx.x;
  if (i < n) d[i] = (__bf16)s[i];
}

// W[K][Nout] (row-major) -> Wt[Nout][K] bf16
__global__ void k_transpose_bf16(const float* __restrict__ W, __bf16* __restrict__ Wt,
                                 int K, int Nout) {
  int i = blockIdx.x * blockDim.x + threadIdx.x;
  if (i < K * Nout) {
    int k = i / Nout, n = i % Nout;
    Wt[(size_t)n * K + k] = (__bf16)W[i];
  }
}

// z[n*ch + c] = b[c]
__global__ void k_bias_init(const float* __restrict__ b, float* __restrict__ z, int n, int ch) {
  int i = blockIdx.x * blockDim.x + threadIdx.x;
  if (i < n) z[i] = b[i % ch];
}

// ---------------- WMMA GEMM: Y[M][Nout] = Xb[M][K] @ Wt[Nout][K]^T ----------------
// bf16 inputs, f32 accumulate. Each wave computes a 16x64 output strip:
// one A fragment per K-step is reused across 4 WMMAs (4x less A traffic).
__global__ void k_gemm_wmma(const __bf16* __restrict__ Xb, const __bf16* __restrict__ Wt,
                            float* __restrict__ Y, int M, int K, int Nout) {
  const int lane = threadIdx.x & 31;
  const int wave = threadIdx.x >> 5;
  const int ntn  = Nout >> 6;            // number of 64-wide strips
  const int tile = blockIdx.x * 4 + wave;
  if (tile >= (M >> 4) * ntn) return;
  const int m0 = (tile / ntn) << 4;
  const int n0 = (tile % ntn) << 6;
  const int half = lane >> 4;            // 0: lanes 0-15, 1: lanes 16-31
  const int l15  = lane & 15;

  // A: row m0+l15, contiguous K (half selects K+16)
  const __bf16* arow = Xb + (size_t)(m0 + l15) * K + half * 16;
  // B: column n0+l15 (+16 per sub-tile), contiguous K (half selects K+16)
  const __bf16* brow = Wt + (size_t)(n0 + l15) * K + half * 16;
  const size_t bstep = (size_t)16 * K;   // next 16 output columns

  v8f acc0 = {}, acc1 = {}, acc2 = {}, acc3 = {};
  for (int k0 = 0; k0 < K; k0 += 32) {
    v8bf alo = *(const v8bf*)(arow + k0);
    v8bf ahi = *(const v8bf*)(arow + k0 + 8);
    v16bf a;
#pragma unroll
    for (int i = 0; i < 8; ++i) { a[i] = alo[i]; a[i + 8] = ahi[i]; }
    v16bf b0 = *(const v16bf*)(brow + k0);
    v16bf b1 = *(const v16bf*)(brow + bstep + k0);
    v16bf b2 = *(const v16bf*)(brow + 2 * bstep + k0);
    v16bf b3 = *(const v16bf*)(brow + 3 * bstep + k0);
    acc0 = __builtin_amdgcn_wmma_f32_16x16x32_bf16(false, a, false, b0, (short)0, acc0, false, false);
    acc1 = __builtin_amdgcn_wmma_f32_16x16x32_bf16(false, a, false, b1, (short)0, acc1, false, false);
    acc2 = __builtin_amdgcn_wmma_f32_16x16x32_bf16(false, a, false, b2, (short)0, acc2, false, false);
    acc3 = __builtin_amdgcn_wmma_f32_16x16x32_bf16(false, a, false, b3, (short)0, acc3, false, false);
  }
  // D layout: vgpr r -> row m0 + r + half*8, lane&15 -> col
  float* yout = Y + (size_t)(m0 + half * 8) * Nout + n0 + l15;
#pragma unroll
  for (int r = 0; r < 8; ++r) {
    yout[(size_t)r * Nout +  0] = acc0[r];
    yout[(size_t)r * Nout + 16] = acc1[r];
    yout[(size_t)r * Nout + 32] = acc2[r];
    yout[(size_t)r * Nout + 48] = acc3[r];
  }
}

// ---------------- edge-phase kernels (GATv2 attention) ----------------

__device__ __forceinline__ unsigned encf(float x) {
  unsigned u = __float_as_uint(x);
  return (u & 0x80000000u) ? ~u : (u | 0x80000000u);
}
__device__ __forceinline__ float decf(unsigned u) {
  return (u & 0x80000000u) ? __uint_as_float(u & 0x7FFFFFFFu) : __uint_as_float(~u);
}

// logits[e,h] = att[h] . leakyrelu(xl[src] + xr[dst]); atomic segment-max over dst
__global__ void k_edge_logits(const float* __restrict__ xl, const float* __restrict__ xr,
                              const float* __restrict__ att,
                              const int* __restrict__ srcA, const int* __restrict__ dstA,
                              float* __restrict__ logits, unsigned* __restrict__ mmax,
                              int heads, int ch) {
  int t = blockIdx.x * blockDim.x + threadIdx.x;
  if (t >= ET * heads) return;
  int e = t / heads, h = t % heads;
  int s = (e < E_EDGES) ? srcA[e] : (e - E_EDGES);
  int d = (e < E_EDGES) ? dstA[e] : (e - E_EDGES);
  int hc = heads * ch;
  const float* pl = xl + (size_t)s * hc + h * ch;
  const float* pr = xr + (size_t)d * hc + h * ch;
  const float* pa = att + h * ch;
  __builtin_prefetch(pl, 0, 3);
  __builtin_prefetch(pr, 0, 3);
  float sum = 0.f;
  for (int c = 0; c < ch; c += 4) {
    v4f a = *(const v4f*)(pl + c);
    v4f b = *(const v4f*)(pr + c);
    v4f w = *(const v4f*)(pa + c);
#pragma unroll
    for (int j = 0; j < 4; ++j) {
      float v = a[j] + b[j];
      v = (v > 0.f) ? v : NEG_SLOPE * v;
      sum += w[j] * v;
    }
  }
  logits[t] = sum;
  atomicMax(&mmax[(size_t)d * heads + h], encf(sum));
}

// ea = exp(logit - max); atomic segment-sum of ea over dst
__global__ void k_edge_exp(const float* __restrict__ logits, const unsigned* __restrict__ mmax,
                           const int* __restrict__ dstA,
                           float* __restrict__ ea, float* __restrict__ denom, int heads) {
  int t = blockIdx.x * blockDim.x + threadIdx.x;
  if (t >= ET * heads) return;
  int e = t / heads, h = t % heads;
  int d = (e < E_EDGES) ? dstA[e] : (e - E_EDGES);
  float m = decf(mmax[(size_t)d * heads + h]);
  float v = __expf(logits[t] - m);
  ea[t] = v;
  atomicAdd(&denom[(size_t)d * heads + h], v);
}

// out[dst] += (ea/denom) * xl[src]
__global__ void k_edge_agg(const float* __restrict__ ea, const float* __restrict__ denom,
                           const float* __restrict__ xl,
                           const int* __restrict__ srcA, const int* __restrict__ dstA,
                           float* __restrict__ out, int heads, int ch) {
  int t = blockIdx.x * blockDim.x + threadIdx.x;
  if (t >= ET * heads) return;
  int e = t / heads, h = t % heads;
  int s = (e < E_EDGES) ? srcA[e] : (e - E_EDGES);
  int d = (e < E_EDGES) ? dstA[e] : (e - E_EDGES);
  int hc = heads * ch;
  float alpha = ea[t] / (denom[(size_t)d * heads + h] + 1e-16f);
  const float* pl = xl + (size_t)s * hc + h * ch;
  float* po = out + (size_t)d * hc + h * ch;
  for (int c = 0; c < ch; c += 4) {
    v4f v = *(const v4f*)(pl + c);
    atomicAdd(&po[c + 0], alpha * v[0]);
    atomicAdd(&po[c + 1], alpha * v[1]);
    atomicAdd(&po[c + 2], alpha * v[2]);
    atomicAdd(&po[c + 3], alpha * v[3]);
  }
}

// ---------------- batch norm (training-mode batch stats) + ReLU -> bf16 ----------------

// column sums + sumsq over rows (coalesced: 256 threads = 256 channels per row)
__global__ void k_bn_stats(const float* __restrict__ x, float* __restrict__ acc) {
  int ch = threadIdx.x;  // blockDim.x == HC
  int rows_per = (N_NODES + gridDim.x - 1) / gridDim.x;
  int r0 = blockIdx.x * rows_per;
  int r1 = min(r0 + rows_per, N_NODES);
  float s = 0.f, s2 = 0.f;
  for (int r = r0; r < r1; ++r) {
    float v = x[(size_t)r * HC + ch];
    s += v; s2 += v * v;
  }
  atomicAdd(&acc[ch], s);
  atomicAdd(&acc[HC + ch], s2);
}

__global__ void k_bn_apply_relu_bf16(const float* __restrict__ x, const float* __restrict__ acc,
                                     const float* __restrict__ gamma, const float* __restrict__ beta,
                                     __bf16* __restrict__ y) {
  int i = blockIdx.x * blockDim.x + threadIdx.x;
  if (i >= N_NODES * HC) return;
  int ch = i & (HC - 1);
  const float inv_n = 1.f / (float)N_NODES;
  float mu  = acc[ch] * inv_n;
  float var = acc[HC + ch] * inv_n - mu * mu;
  float v = gamma[ch] * (x[i] - mu) * rsqrtf(var + BN_EPS) + beta[ch];
  y[i] = (__bf16)(v > 0.f ? v : 0.f);
}

// ---------------- graph mean pooling ----------------

__global__ void k_pool_cnt(const int* __restrict__ batch, float* __restrict__ cnt) {
  int n = blockIdx.x * blockDim.x + threadIdx.x;
  if (n < N_NODES) atomicAdd(&cnt[batch[n]], 1.f);
}

__global__ void k_pool_sum(const float* __restrict__ z, const int* __restrict__ batch,
                           float* __restrict__ g) {
  int t = blockIdx.x * blockDim.x + threadIdx.x;
  if (t >= N_NODES * CH) return;
  int n = t >> 6, c = t & 63;
  atomicAdd(&g[(size_t)batch[n] * CH + c], z[t]);
}

__global__ void k_pool_div(float* __restrict__ g, const float* __restrict__ cnt) {
  int i = blockIdx.x * blockDim.x + threadIdx.x;
  if (i < G_GRAPHS * CH) g[i] /= fmaxf(cnt[i / CH], 1.f);
}

// ---------------- host side ----------------

static inline int T256(long long n) { return (int)((n + 255) / 256); }

static void run_edges(const float* xl, const float* xr, const float* att,
                      const int* src, const int* dst,
                      float* logits, float* ea, unsigned* mmax, float* denom,
                      float* dest, int heads, int ch, bool zero_dest,
                      hipStream_t stream) {
  long long nt = (long long)ET * heads;
  k_fill_u32<<<T256(N_NODES * heads), 256, 0, stream>>>(mmax, ENC_NEG_INF, N_NODES * heads);
  k_fill_f32<<<T256(N_NODES * heads), 256, 0, stream>>>(denom, 0.f, N_NODES * heads);
  if (zero_dest)
    k_fill_f32<<<T256((long long)N_NODES * heads * ch), 256, 0, stream>>>(
        dest, 0.f, N_NODES * heads * ch);
  k_edge_logits<<<T256(nt), 256, 0, stream>>>(xl, xr, att, src, dst, logits, mmax, heads, ch);
  k_edge_exp<<<T256(nt), 256, 0, stream>>>(logits, mmax, dst, ea, denom, heads);
  k_edge_agg<<<T256(nt), 256, 0, stream>>>(ea, denom, xl, src, dst, dest, heads, ch);
}

static void run_gemm(const __bf16* xb, const __bf16* wt, float* y,
                     int M, int K, int Nout, hipStream_t stream) {
  int tiles = (M >> 4) * (Nout >> 6);
  k_gemm_wmma<<<(tiles + 3) / 4, 128, 0, stream>>>(xb, wt, y, M, K, Nout);
}

extern "C" void kernel_launch(void* const* d_in, const int* in_sizes, int n_in,
                              void* d_out, int out_size, void* d_ws, size_t ws_size,
                              hipStream_t stream) {
  const float* x     = (const float*)d_in[0];
  const int*   eidx  = (const int*)d_in[1];
  const int*   batch = (const int*)d_in[2];
  const float* Wl1 = (const float*)d_in[3],  *Wr1 = (const float*)d_in[4];
  const float* att1= (const float*)d_in[5];
  const float* g1  = (const float*)d_in[7],  *be1 = (const float*)d_in[8];
  const float* Wl2 = (const float*)d_in[9],  *Wr2 = (const float*)d_in[10];
  const float* att2= (const float*)d_in[11];
  const float* g2  = (const float*)d_in[13], *be2 = (const float*)d_in[14];
  const float* Wl3 = (const float*)d_in[15], *Wr3 = (const float*)d_in[16];
  const float* att3= (const float*)d_in[17];
  const float* b3  = (const float*)d_in[18];
  const int* src = eidx;
  const int* dst = eidx + E_EDGES;

  char* ws = (char*)d_ws;
  size_t off = 0;
  auto alloc = [&](size_t bytes) { size_t c = off; off += (bytes + 255) & ~(size_t)255; return c; };
  __bf16*   xb     = (__bf16*)(ws + alloc((size_t)N_NODES * HC * 2));
  __bf16*   wtl    = (__bf16*)(ws + alloc((size_t)HC * HC * 2));
  __bf16*   wtr    = (__bf16*)(ws + alloc((size_t)HC * HC * 2));
  float*    xl     = (float*)(ws + alloc((size_t)N_NODES * HC * 4));
  float*    xr     = (float*)(ws + alloc((size_t)N_NODES * HC * 4));
  float*    agg    = (float*)(ws + alloc((size_t)N_NODES * HC * 4));
  float*    logits = (float*)(ws + alloc((size_t)ET * HEADS * 4));
  float*    ea     = (float*)(ws + alloc((size_t)ET * HEADS * 4));
  unsigned* mmax   = (unsigned*)(ws + alloc((size_t)N_NODES * HEADS * 4));
  float*    denom  = (float*)(ws + alloc((size_t)N_NODES * HEADS * 4));
  float*    bnacc  = (float*)(ws + alloc(2 * HC * 4));
  float*    cnt    = (float*)(ws + alloc(G_GRAPHS * 4));
  (void)ws_size; (void)n_in; (void)in_sizes; (void)out_size;

  float* z    = (float*)d_out;            // [N, CH]
  float* gout = z + (size_t)N_NODES * CH; // [G, CH]

  // ---------- Layer 1 (IN -> H*C) ----------
  k_f32_to_bf16<<<T256((long long)N_NODES * IN_CH), 256, 0, stream>>>(x, xb, N_NODES * IN_CH);
  k_transpose_bf16<<<T256(IN_CH * HC), 256, 0, stream>>>(Wl1, wtl, IN_CH, HC);
  k_transpose_bf16<<<T256(IN_CH * HC), 256, 0, stream>>>(Wr1, wtr, IN_CH, HC);
  run_gemm(xb, wtl, xl, N_NODES, IN_CH, HC, stream);
  run_gemm(xb, wtr, xr, N_NODES, IN_CH, HC, stream);
  run_edges(xl, xr, att1, src, dst, logits, ea, mmax, denom, agg, HEADS, CH, true, stream);
  // BN + ReLU (b1 cancels under batch-stat BN) -> bf16 input for next layer
  k_fill_f32<<<T256(2 * HC), 256, 0, stream>>>(bnacc, 0.f, 2 * HC);
  k_bn_stats<<<256, HC, 0, stream>>>(agg, bnacc);
  k_bn_apply_relu_bf16<<<T256((long long)N_NODES * HC), 256, 0, stream>>>(agg, bnacc, g1, be1, xb);

  // ---------- Layer 2 (H*C -> H*C) ----------
  k_transpose_bf16<<<T256(HC * HC), 256, 0, stream>>>(Wl2, wtl, HC, HC);
  k_transpose_bf16<<<T256(HC * HC), 256, 0, stream>>>(Wr2, wtr, HC, HC);
  run_gemm(xb, wtl, xl, N_NODES, HC, HC, stream);
  run_gemm(xb, wtr, xr, N_NODES, HC, HC, stream);
  run_edges(xl, xr, att2, src, dst, logits, ea, mmax, denom, agg, HEADS, CH, true, stream);
  k_fill_f32<<<T256(2 * HC), 256, 0, stream>>>(bnacc, 0.f, 2 * HC);
  k_bn_stats<<<256, HC, 0, stream>>>(agg, bnacc);
  k_bn_apply_relu_bf16<<<T256((long long)N_NODES * HC), 256, 0, stream>>>(agg, bnacc, g2, be2, xb);

  // ---------- Layer 3 (H*C -> C, heads=1) ----------
  k_transpose_bf16<<<T256(HC * CH), 256, 0, stream>>>(Wl3, wtl, HC, CH);
  k_transpose_bf16<<<T256(HC * CH), 256, 0, stream>>>(Wr3, wtr, HC, CH);
  run_gemm(xb, wtl, xl, N_NODES, HC, CH, stream);
  run_gemm(xb, wtr, xr, N_NODES, HC, CH, stream);
  // z initialized with bias b3; aggregation accumulates on top
  k_bias_init<<<T256((long long)N_NODES * CH), 256, 0, stream>>>(b3, z, N_NODES * CH, CH);
  run_edges(xl, xr, att3, src, dst, logits, ea, mmax, denom, z, 1, CH, false, stream);

  // ---------- graph mean pooling ----------
  k_fill_f32<<<T256(G_GRAPHS * CH), 256, 0, stream>>>(gout, 0.f, G_GRAPHS * CH);
  k_fill_f32<<<T256(G_GRAPHS), 256, 0, stream>>>(cnt, 0.f, G_GRAPHS);
  k_pool_cnt<<<T256(N_NODES), 256, 0, stream>>>(batch, cnt);
  k_pool_sum<<<T256((long long)N_NODES * CH), 256, 0, stream>>>(z, batch, gout);
  k_pool_div<<<T256(G_GRAPHS * CH), 256, 0, stream>>>(gout, cnt);
}
